// ST_SatMVS_9758165696780
// MI455X (gfx1250) — matrix operand order repaired and checked by reading, not compiled, not run on hardware
//
#include <hip/hip_runtime.h>
#include <hip/hip_bf16.h>
#include <stdint.h>

typedef __attribute__((ext_vector_type(16))) _Float16 v16h;
typedef __attribute__((ext_vector_type(8)))  _Float16 v8h;
typedef __attribute__((ext_vector_type(8)))  float    v8f;

#define Vn  3
#define Cc  32
#define Hh  128
#define Ww  128
#define Dd  48
#define HW_  (Hh * Ww)
#define DHW_ (Dd * HW_)
#define Dp  (Dd + 2)
#define Hp  (Hh + 2)
#define Wp  (Ww + 2)

// ---------------------------------------------------------------------------
// 1. proj = src_proj @ inv(ref_proj); extract rot (3x3) + trans (3) per view.
//    ref_proj last row is [0,0,0,1] -> inv([A b;0 1]) = [A^-1 -A^-1 b; 0 1].
// ---------------------------------------------------------------------------
__global__ void k_proj(const float* __restrict__ projs, float* __restrict__ rt) {
  const float* R = projs;  // ref (view 0) 4x4, row-major
  float a00 = R[0], a01 = R[1], a02 = R[2],  b0 = R[3];
  float a10 = R[4], a11 = R[5], a12 = R[6],  b1 = R[7];
  float a20 = R[8], a21 = R[9], a22 = R[10], b2 = R[11];
  float det = a00 * (a11 * a22 - a12 * a21)
            - a01 * (a10 * a22 - a12 * a20)
            + a02 * (a10 * a21 - a11 * a20);
  float id = 1.0f / det;
  float i00 =  (a11 * a22 - a12 * a21) * id;
  float i01 = -(a01 * a22 - a02 * a21) * id;
  float i02 =  (a01 * a12 - a02 * a11) * id;
  float i10 = -(a10 * a22 - a12 * a20) * id;
  float i11 =  (a00 * a22 - a02 * a20) * id;
  float i12 = -(a00 * a12 - a02 * a10) * id;
  float i20 =  (a10 * a21 - a11 * a20) * id;
  float i21 = -(a00 * a21 - a01 * a20) * id;
  float i22 =  (a00 * a11 - a01 * a10) * id;
  float c0 = -(i00 * b0 + i01 * b1 + i02 * b2);
  float c1 = -(i10 * b0 + i11 * b1 + i12 * b2);
  float c2 = -(i20 * b0 + i21 * b1 + i22 * b2);
  float inv[16] = { i00, i01, i02, c0,
                    i10, i11, i12, c1,
                    i20, i21, i22, c2,
                    0.f, 0.f, 0.f, 1.f };
  for (int v = 1; v < Vn; ++v) {
    const float* S = projs + v * 16;
    float* o = rt + (v - 1) * 12;
    for (int r = 0; r < 3; ++r) {
      for (int col = 0; col < 3; ++col) {
        float s = 0.f;
        for (int k = 0; k < 4; ++k) s += S[r * 4 + k] * inv[k * 4 + col];
        o[r * 3 + col] = s;
      }
      float s = 0.f;
      for (int k = 0; k < 4; ++k) s += S[r * 4 + k] * inv[k * 4 + 3];
      o[9 + r] = s;
    }
  }
}

// ---------------------------------------------------------------------------
// 2. Pack w1 [8,32,3,3,3] into per-tap WMMA A-fragments (16x32 f16, M padded
//    to 16). A-layout: lane l holds row m=l&15; low-half lanes hold
//    K={0..15}, high-half lanes K={16..31} (pairs per VGPR).
// ---------------------------------------------------------------------------
__global__ void k_pack_w1(const float* __restrict__ w1, _Float16* __restrict__ Apack) {
  int t    = blockIdx.x;       // tap 0..26 (kd*9+kh*3+kw)
  int lane = threadIdx.x;      // 0..31
  int m     = lane & 15;
  int kHalf = (lane >> 4) * 16;
  v16h frag;
  #pragma unroll
  for (int e = 0; e < 16; ++e) {
    int c = kHalf + e;
    float val = (m < 8) ? w1[(m * Cc + c) * 27 + t] : 0.0f;
    frag[e] = (_Float16)val;
  }
  *(v16h*)(Apack + (t * 32 + lane) * 16) = frag;
}

// ---------------------------------------------------------------------------
// 3. Zero fill (halo => SAME zero padding for the convs).
// ---------------------------------------------------------------------------
__global__ void k_zero(uint32_t* __restrict__ p, int n) {
  int i = blockIdx.x * blockDim.x + threadIdx.x;
  if (i < n) p[i] = 0u;
}

// ---------------------------------------------------------------------------
// 4. Homography warp + bilinear (zeros padding) + 3-view variance.
//    Writes f16 variance volume, layout [d+1][h+1][w+1][c] (channels inner).
// ---------------------------------------------------------------------------
__global__ void k_variance(const float* __restrict__ feat,
                           const float* __restrict__ rt,
                           const float* __restrict__ depthv,
                           _Float16* __restrict__ var) {
  int tid = blockIdx.x * blockDim.x + threadIdx.x;
  if (tid >= DHW_) return;
  int w = tid & (Ww - 1);
  int h = (tid >> 7) & (Hh - 1);
  int d = tid >> 14;

  float dep = depthv[tid];
  float x = (float)w, y = (float)h;

  int   off[2][4];
  float wt[2][4];
  #pragma unroll
  for (int v = 0; v < 2; ++v) {
    const float* m = rt + v * 12;
    float X = (m[0] * x + m[1] * y + m[2]) * dep + m[9];
    float Y = (m[3] * x + m[4] * y + m[5]) * dep + m[10];
    float Z = (m[6] * x + m[7] * y + m[8]) * dep + m[11];
    float iz = 1.0f / Z;
    float px = X * iz, py = Y * iz;
    float x0f = floorf(px), y0f = floorf(py);
    float wx = px - x0f, wy = py - y0f;
    int x0 = (int)x0f, y0 = (int)y0f;
    int x1 = x0 + 1,  y1 = y0 + 1;
    bool vx0 = (x0 >= 0) & (x0 <= Ww - 1), vx1 = (x1 >= 0) & (x1 <= Ww - 1);
    bool vy0 = (y0 >= 0) & (y0 <= Hh - 1), vy1 = (y1 >= 0) & (y1 <= Hh - 1);
    int xc0 = min(max(x0, 0), Ww - 1), xc1 = min(max(x1, 0), Ww - 1);
    int yc0 = min(max(y0, 0), Hh - 1), yc1 = min(max(y1, 0), Hh - 1);
    wt[v][0] = (vx0 & vy0) ? (1.f - wx) * (1.f - wy) : 0.f;
    wt[v][1] = (vx1 & vy0) ? wx * (1.f - wy)         : 0.f;
    wt[v][2] = (vx0 & vy1) ? (1.f - wx) * wy         : 0.f;
    wt[v][3] = (vx1 & vy1) ? wx * wy                 : 0.f;
    off[v][0] = yc0 * Ww + xc0;
    off[v][1] = yc0 * Ww + xc1;
    off[v][2] = yc1 * Ww + xc0;
    off[v][3] = yc1 * Ww + xc1;
  }

  const float* f1 = feat + 1 * Cc * HW_;
  const float* f2 = feat + 2 * Cc * HW_;
  int pix = h * Ww + w;

  alignas(16) _Float16 outv[Cc];
  #pragma unroll 4
  for (int c = 0; c < Cc; ++c) {
    float r  = feat[c * HW_ + pix];
    float s  = r;
    float sq = r * r;
    const float* p1 = f1 + c * HW_;
    float val = wt[0][0] * p1[off[0][0]] + wt[0][1] * p1[off[0][1]]
              + wt[0][2] * p1[off[0][2]] + wt[0][3] * p1[off[0][3]];
    s += val; sq += val * val;
    const float* p2 = f2 + c * HW_;
    val = wt[1][0] * p2[off[1][0]] + wt[1][1] * p2[off[1][1]]
        + wt[1][2] * p2[off[1][2]] + wt[1][3] * p2[off[1][3]];
    s += val; sq += val * val;
    float mean = s * (1.0f / 3.0f);
    outv[c] = (_Float16)(sq * (1.0f / 3.0f) - mean * mean);
  }

  _Float16* dst = var + ((long)((d + 1) * Hp + (h + 1)) * Wp + (w + 1)) * Cc;
  float4* d4 = (float4*)dst;
  const float4* s4 = (const float4*)outv;
  d4[0] = s4[0]; d4[1] = s4[1]; d4[2] = s4[2]; d4[3] = s4[3];
}

// ---------------------------------------------------------------------------
// 5. Conv3D 32->8 as implicit GEMM on WMMA f16: M=16 (8 real out-ch), N=16
//    positions/tile, K=32/tap. Each wave: 4 tiles (64 w-positions) x 27 taps
//    = 108 v_wmma, fully unrolled so every tap's address is a compile-time
//    IOFFSET off one base pointer (no per-tap scalar math).
// ---------------------------------------------------------------------------
__device__ __forceinline__ void store_h(v8f acc, const float* bias, _Float16* dst) {
  alignas(16) _Float16 hv[8];
  #pragma unroll
  for (int j = 0; j < 8; ++j) {
    float v = acc[j] + bias[j];
    hv[j] = (_Float16)fmaxf(v, 0.0f);
  }
  *(float4*)dst = *(const float4*)hv;
}

__global__ void k_conv1_wmma(const _Float16* __restrict__ var,
                             const _Float16* __restrict__ Apack,
                             const float* __restrict__ bias1,
                             _Float16* __restrict__ hbuf) {
  int wave = threadIdx.x >> 5;
  int lane = threadIdx.x & 31;
  int seg  = blockIdx.x * 4 + wave;   // 64-position segment
  int pos0 = seg * 64;
  int w0 = pos0 & (Ww - 1);
  int h  = (pos0 >> 7) & (Hh - 1);
  int d  = pos0 >> 14;

  int n    = lane & 15;          // column (position within tile)
  int kOff = (lane >> 4) * 16;   // K half handled by this lane

  v8f acc0 = {}, acc1 = {}, acc2 = {}, acc3 = {};

  // One base pointer; all 27 tap offsets become compile-time constants that
  // fold into global_load IOFFSET immediates (max ~2.2MB < 2^23).
  const _Float16* base = var + ((long)(d * Hp + h) * Wp + (w0 + n)) * Cc + kOff;
  const _Float16* apL  = Apack + lane * 16;

  #pragma unroll
  for (int kd = 0; kd < 3; ++kd)
    #pragma unroll
    for (int kh = 0; kh < 3; ++kh)
      #pragma unroll
      for (int kw = 0; kw < 3; ++kw) {
        const int t = (kd * 3 + kh) * 3 + kw;
        const _Float16* rp = base + ((kd * Hp + kh) * Wp + kw) * Cc;
        v16h a   = *(const v16h*)(apL + t * 512);       // 32 lanes * 16 halfs
        v16h bv0 = *(const v16h*)(rp);
        v16h bv1 = *(const v16h*)(rp + 16 * Cc);
        v16h bv2 = *(const v16h*)(rp + 32 * Cc);
        v16h bv3 = *(const v16h*)(rp + 48 * Cc);
        acc0 = __builtin_amdgcn_wmma_f32_16x16x32_f16(false, a, false, bv0, (short)0, acc0, false, false);
        acc1 = __builtin_amdgcn_wmma_f32_16x16x32_f16(false, a, false, bv1, (short)0, acc1, false, false);
        acc2 = __builtin_amdgcn_wmma_f32_16x16x32_f16(false, a, false, bv2, (short)0, acc2, false, false);
        acc3 = __builtin_amdgcn_wmma_f32_16x16x32_f16(false, a, false, bv3, (short)0, acc3, false, false);
      }

  float bias[8];
  #pragma unroll
  for (int j = 0; j < 8; ++j) bias[j] = bias1[j];

  if (lane < 16) {   // lanes 0-15 hold out-channels 0..7 for position n
    long obase = (long)((d + 1) * Hp + (h + 1)) * Wp + 1;
    store_h(acc0, bias, hbuf + (obase + w0 + 0  + n) * 8);
    store_h(acc1, bias, hbuf + (obase + w0 + 16 + n) * 8);
    store_h(acc2, bias, hbuf + (obase + w0 + 32 + n) * 8);
    store_h(acc3, bias, hbuf + (obase + w0 + 48 + n) * 8);
  }
}

// ---------------------------------------------------------------------------
// 6. Conv3D 8->1 (K=216): weights in LDS, hidden volume read as 16B vectors.
// ---------------------------------------------------------------------------
__global__ void k_conv2(const _Float16* __restrict__ hbuf,
                        const float* __restrict__ w2,
                        const float* __restrict__ b2,
                        float* __restrict__ logits) {
  __shared__ float sw[216];
  __shared__ float sb;
  if (threadIdx.x < 216) sw[threadIdx.x] = w2[threadIdx.x];
  if (threadIdx.x == 0)  sb = b2[0];
  __syncthreads();

  int tid = blockIdx.x * blockDim.x + threadIdx.x;
  if (tid >= DHW_) return;
  int w = tid & (Ww - 1);
  int h = (tid >> 7) & (Hh - 1);
  int d = tid >> 14;

  const _Float16* base = hbuf + ((long)(d * Hp + h) * Wp + w) * 8;
  float acc = sb;
  #pragma unroll
  for (int kd = 0; kd < 3; ++kd)
    #pragma unroll
    for (int kh = 0; kh < 3; ++kh)
      #pragma unroll
      for (int kw = 0; kw < 3; ++kw) {
        int t = (kd * 3 + kh) * 3 + kw;
        v8h hv = *(const v8h*)(base + ((kd * Hp + kh) * Wp + kw) * 8);
        #pragma unroll
        for (int j = 0; j < 8; ++j) acc += (float)hv[j] * sw[j * 27 + t];
      }
  logits[tid] = acc;
}

// ---------------------------------------------------------------------------
// 7. Online softmax over D + depth regression; max-prob = 1/sum(exp(x-max)).
// ---------------------------------------------------------------------------
__global__ void k_softmax_depth(const float* __restrict__ logits,
                                const float* __restrict__ depthv,
                                float* __restrict__ out) {
  int p = blockIdx.x * blockDim.x + threadIdx.x;
  if (p >= HW_) return;
  float m = -1e30f, s = 0.f, td = 0.f;
  for (int d = 0; d < Dd; ++d) {
    float x  = logits[d * HW_ + p];
    float dv = depthv[d * HW_ + p];
    float mn = fmaxf(m, x);
    float sc = __expf(m - mn);
    float e  = __expf(x - mn);
    s  = s * sc + e;
    td = td * sc + e * dv;
    m = mn;
  }
  out[p]       = td / s;     // expected depth
  out[HW_ + p] = 1.0f / s;   // photometric confidence (max prob)
}

// ---------------------------------------------------------------------------
extern "C" void kernel_launch(void* const* d_in, const int* in_sizes, int n_in,
                              void* d_out, int out_size, void* d_ws, size_t ws_size,
                              hipStream_t stream) {
  (void)in_sizes; (void)n_in; (void)out_size; (void)ws_size;
  const float* features = (const float*)d_in[0];
  const float* projs    = (const float*)d_in[1];
  const float* depthv   = (const float*)d_in[2];
  const float* w1       = (const float*)d_in[3];
  const float* b1       = (const float*)d_in[4];
  const float* w2       = (const float*)d_in[5];
  const float* b2       = (const float*)d_in[6];
  float* out = (float*)d_out;

  // Workspace layout (all 256B aligned)
  constexpr size_t VAR_ELEMS   = (size_t)Dp * Hp * Wp * Cc;    // 27,040,000 f16
  constexpr size_t VAR_BYTES   = VAR_ELEMS * 2;                // 54,080,000
  constexpr size_t APACK_BYTES = 27 * 32 * 16 * 2;             // 27,648
  constexpr size_t H_ELEMS     = (size_t)Dp * Hp * Wp * 8;     // 6,760,000 f16
  constexpr size_t H_BYTES     = ((H_ELEMS * 2 + 255) / 256) * 256;
  constexpr size_t LOG_BYTES   = (size_t)DHW_ * 4;

  char* ws = (char*)d_ws;
  _Float16* var    = (_Float16*)ws;
  _Float16* Apack  = (_Float16*)(ws + VAR_BYTES);
  _Float16* hbuf   = (_Float16*)(ws + VAR_BYTES + APACK_BYTES);
  float*    logits = (float*)   (ws + VAR_BYTES + APACK_BYTES + H_BYTES);
  float*    rt     = (float*)   (ws + VAR_BYTES + APACK_BYTES + H_BYTES + LOG_BYTES);

  // 1. projection matrices
  k_proj<<<1, 1, 0, stream>>>(projs, rt);
  // 2. weight packing for WMMA A-fragments
  k_pack_w1<<<27, 32, 0, stream>>>(w1, Apack);
  // 3. zero halo buffers
  {
    int nv = (int)(VAR_ELEMS / 2);            // f16 pairs as u32
    k_zero<<<(nv + 255) / 256, 256, 0, stream>>>((uint32_t*)var, nv);
    int nh = (int)(H_ELEMS / 2);
    k_zero<<<(nh + 255) / 256, 256, 0, stream>>>((uint32_t*)hbuf, nh);
  }
  // 4. warp + variance volume (f16, channels innermost)
  k_variance<<<(DHW_ + 255) / 256, 256, 0, stream>>>(features, rt, depthv, var);
  // 5. conv1 via WMMA implicit GEMM: DHW/64 waves, 4 waves/block
  k_conv1_wmma<<<DHW_ / 256, 128, 0, stream>>>(var, Apack, b1, hbuf);
  // 6. conv2
  k_conv2<<<(DHW_ + 255) / 256, 256, 0, stream>>>(hbuf, w2, b2, logits);
  // 7. softmax + depth regression
  k_softmax_depth<<<(HW_ + 255) / 256, 256, 0, stream>>>(logits, depthv, out);
}
